// ExpertScatter_37117107372440
// MI455X (gfx1250) — matrix operand order repaired and checked by reading, not compiled, not run on hardware
//
#include <hip/hip_runtime.h>

typedef __attribute__((ext_vector_type(16))) __bf16 v16bf;
typedef __attribute__((ext_vector_type(8)))  float  v8f;

constexpr int BATCH = 8;
constexpr int HEADS = 16;
constexpr int KTOK  = 1024;   // tokens per (b, head)  == GEMM M dim
constexpr int HDIM  = 128;    // reduction dim
constexpr int ODIM  = 1024;   // GEMM N dim
constexpr int TSLOT = 4096;   // scatter slots

// Swizzled split-W layout: index = (((e*4 + kt)*64 + c)*32 + lane)*16 bf16
// kt = 32-wide K tile (HDIM/32 = 4), c = 16-wide N tile (ODIM/16 = 64).
constexpr long WSPLIT_ELEMS = (long)HEADS * HDIM * ODIM;   // 2,097,152 per matrix

// ---------------------------------------------------------------------------
// Zero-fill the 134 MB output (harness poisons it with 0xAA before timing).
// ---------------------------------------------------------------------------
__global__ __launch_bounds__(256) void es_zero_kernel(float4* __restrict__ out, long n4) {
  long i = (long)blockIdx.x * blockDim.x + threadIdx.x;
  if (i < n4) out[i] = make_float4(0.f, 0.f, 0.f, 0.f);
}

// ---------------------------------------------------------------------------
// One-time: split fp32 W into bf16 hi/lo, stored pre-swizzled so a WMMA
// B-fragment (32 K x 16 N, bf16) is 32 contiguous bytes per lane.
// B layout (this code's contiguous k order): lanes 0-15 -> K=kbase+0..15, lanes 16-31 ->
// K=kbase+16..31 ... here kbase=(lane/16)*16 within the 32-wide K tile;
// VGPR v holds K = {2v, 2v+1} packed -> element q maps to K = kbase+q.
// ---------------------------------------------------------------------------
__global__ __launch_bounds__(256) void es_wsplit_kernel(
    const float* __restrict__ W,      // [HEADS, HDIM, ODIM] fp32
    __bf16* __restrict__ Wh, __bf16* __restrict__ Wl)
{
  const int t    = blockIdx.x * 256 + threadIdx.x;   // 0 .. 131071
  const int lane = t & 31;
  const int c    = (t >> 5) & 63;
  const int kt   = (t >> 11) & 3;
  const int e    = t >> 13;

  const int n     = c * 16 + (lane & 15);
  const int kbase = kt * 32 + (lane >> 4) * 16;
  const float* wp = W + (long)e * HDIM * ODIM + n;

  v16bf hi, lo;
#pragma unroll
  for (int q = 0; q < 16; ++q) {
    float w  = wp[(long)(kbase + q) * ODIM];
    __bf16 h = (__bf16)w;                 // RNE truncate
    __bf16 l = (__bf16)(w - (float)h);    // residual
    hi[q] = h;
    lo[q] = l;
  }
  const long o = (long)t * 16;
  *(v16bf*)(Wh + o) = hi;                 // coalesced 32B per thread
  *(v16bf*)(Wl + o) = lo;
}

// ---------------------------------------------------------------------------
// Fused per-head GEMM (bf16 WMMA 16x16x32, split-precision 3-term) + atomic
// scatter-add. One wave computes a 16(row) x 64(col) tile of
// X[b,e] = Y[b,e] * W[e], then scatter-adds each row into
// Out[b, Ind[b,e,row], :] with global_atomic_add_f32 (no-return).
// Grid covers tiles exactly -> EXEC all-1s (WMMA requirement), no branches.
// ---------------------------------------------------------------------------
__global__ __launch_bounds__(256) void expert_gemm_scatter(
    const float*  __restrict__ Y,     // [B, H, KTOK, HDIM] fp32
    const int*    __restrict__ Ind,   // [B, H, KTOK]
    const __bf16* __restrict__ Wh,    // swizzled bf16 hi
    const __bf16* __restrict__ Wl,    // swizzled bf16 lo
    float*        __restrict__ Out)   // [B, TSLOT, ODIM] fp32
{
  const int lane = threadIdx.x & 31;
  const int wid  = threadIdx.x >> 5;                  // 8 waves / block
  const long gw  = (long)blockIdx.x * 8 + wid;        // global wave-tile id

  const int colTile = (int)(gw & 15);                 // 16 col tiles of 64
  const int rowTile = (int)((gw >> 4) & 63);          // 64 row tiles of 16
  const int be      = (int)(gw >> 10);                // 0..127  (b*HEADS+e)
  const int e = be & (HEADS - 1);
  const int b = be >> 4;

  const float*  Yp  = Y   + ((long)be * KTOK + (long)rowTile * 16) * HDIM;
  const int*    Ip  = Ind + (long)be * KTOK + rowTile * 16;
  float*        Op  = Out + (long)b * TSLOT * ODIM + colTile * 64;
  const __bf16* WhP = Wh + (long)e * (4 * 64 * 32 * 16);
  const __bf16* WlP = Wl + (long)e * (4 * 64 * 32 * 16);

  // A fragment (matches B): lane m=lane%16; lane<16 -> K=0..15,
  // lane>=16 -> K=16..31  => ksub=(lane/16)*16, one run of 16.
  const int m    = lane & 15;
  const int ksub = (lane >> 4) << 4;

  v8f acc[4] = {v8f{0.f}, v8f{0.f}, v8f{0.f}, v8f{0.f}};

  const float* ya = Yp + (long)m * HDIM + ksub;

#pragma unroll
  for (int kt = 0; kt < HDIM / 32; ++kt) {
    // ---- build A hi/lo fragments from exact fp32 Y (in-register split) ----
    const float* yk = ya + kt * 32;
    v16bf Ah, Al;
#pragma unroll
    for (int q = 0; q < 8; ++q) {
      float y0 = yk[q];            // contiguous 32B run -> b128 loads
      float y1 = yk[8 + q];        // second contiguous 32B run
      __bf16 h0 = (__bf16)y0, h1 = (__bf16)y1;
      Ah[q]     = h0;  Al[q]     = (__bf16)(y0 - (float)h0);
      Ah[8 + q] = h1;  Al[8 + q] = (__bf16)(y1 - (float)h1);
    }

    // ---- 4 B fragments (hi+lo), pre-swizzled: 32B contiguous per lane ----
    const long bbase = (((long)kt * 64 + colTile * 4) * 32 + lane) * 16;
#pragma unroll
    for (int t = 0; t < 4; ++t) {
      v16bf Bh = *(const v16bf*)(WhP + bbase + (long)t * 32 * 16);
      v16bf Bl = *(const v16bf*)(WlP + bbase + (long)t * 32 * 16);
      // split-precision: X = Yh*Wh + Yh*Wl + Yl*Wh  (fp32 accumulate)
      acc[t] = __builtin_amdgcn_wmma_f32_16x16x32_bf16(
          false, Ah, false, Bh, (short)0, acc[t], false, false);
      acc[t] = __builtin_amdgcn_wmma_f32_16x16x32_bf16(
          false, Ah, false, Bl, (short)0, acc[t], false, false);
      acc[t] = __builtin_amdgcn_wmma_f32_16x16x32_bf16(
          false, Al, false, Bh, (short)0, acc[t], false, false);
    }
  }

  // C/D layout: VGPR v, lane l -> M = v + 8*(l>=16), N = l%16
  const int mHi = (lane >> 4) << 3;
#pragma unroll
  for (int v = 0; v < 8; ++v) {
    const int M   = mHi + v;
    const int ind = Ip[M];                       // scatter target slot
    float* rowp = Op + (long)ind * ODIM + m;
#pragma unroll
    for (int t = 0; t < 4; ++t) {
      // lowers to global_atomic_add_f32 (no return) -> resolves in L2
      unsafeAtomicAdd(rowp + t * 16, acc[t][v]);
    }
  }
}

extern "C" void kernel_launch(void* const* d_in, const int* in_sizes, int n_in,
                              void* d_out, int out_size, void* d_ws, size_t ws_size,
                              hipStream_t stream) {
  // setup_inputs order: Y, Ind, T (scalar), W
  const float* Y   = (const float*)d_in[0];
  const int*   Ind = (const int*)d_in[1];
  const float* W   = (const float*)d_in[3];
  float* out = (float*)d_out;

  __bf16* Wh = (__bf16*)d_ws;                 // 4 MB
  __bf16* Wl = Wh + WSPLIT_ELEMS;             // +4 MB  (needs ws >= 8 MB)

  // 1) zero output accumulator
  const long n  = (long)BATCH * TSLOT * ODIM;   // 33.5M floats
  const long n4 = n >> 2;
  es_zero_kernel<<<(int)((n4 + 255) / 256), 256, 0, stream>>>((float4*)out, n4);

  // 2) split + swizzle W into bf16 hi/lo fragments (131072 threads)
  es_wsplit_kernel<<<512, 256, 0, stream>>>(W, Wh, Wl);

  // 3) fused bf16x3 WMMA GEMM + scatter-add
  // wave-tiles = B*H * (KTOK/16) * (ODIM/64) = 128*64*16 = 131072 -> /8 waves
  const int blocks = (BATCH * HEADS * (KTOK / 16) * (ODIM / 64)) / 8; // 16384
  expert_gemm_scatter<<<blocks, 256, 0, stream>>>(Y, Ind, Wh, Wl, out);
}
